// SPFNSMultiHeadAttention_69913477644291
// MI455X (gfx1250) — matrix operand order repaired and checked by reading, not compiled, not run on hardware
//
#include <hip/hip_runtime.h>
#include <math.h>

typedef __bf16 bf16_t;
typedef __attribute__((ext_vector_type(16))) __bf16 v16bf;
typedef __attribute__((ext_vector_type(8)))  __bf16 v8bf;
typedef __attribute__((ext_vector_type(8)))  float  v8f;

namespace {
constexpr int kB  = 4;
constexpr int kS  = 1024;
constexpr int kH  = 16;
constexpr int kDH = 64;
constexpr int kD  = 1024;          // kH * kDH
constexpr int kM  = kB * kS;       // 4096 tokens
constexpr float kClip   = 1.0f - 1e-6f;
constexpr float kEps    = 1e-12f;
constexpr float kPowExp = -65.0f;  // -(d_intrinsic + alpha) = -(64 + 1)
}

#define ZERO8F {0.f, 0.f, 0.f, 0.f, 0.f, 0.f, 0.f, 0.f}

// ---------------------------------------------------------------------------
// WMMA fragment helpers. Pointers are pre-resolved per lane; the two 16-byte
// loads sit at constant element offsets so they fold into the 24-bit
// global_load immediate field (keeps only ONE live 64-bit address per stream).
// ---------------------------------------------------------------------------
__device__ __forceinline__ v16bf make_v16(v8bf lo, v8bf hi) {
  return __builtin_shufflevector(lo, hi, 0, 1, 2, 3, 4, 5, 6, 7,
                                         8, 9, 10, 11, 12, 13, 14, 15);
}

// Generic two-chunk fragment load. For an A fragment pass (k0, k0+16) with
// p = row + half*8; for a B fragment pass (k0, k0+16) with p = row + half*8.
__device__ __forceinline__ v16bf load_frag2(const bf16_t* __restrict__ p,
                                            int off_lo, int off_hi) {
  v8bf lo = *reinterpret_cast<const v8bf*>(p + off_lo);
  v8bf hi = *reinterpret_cast<const v8bf*>(p + off_hi);
  return make_v16(lo, hi);
}

#define WMMA_BF16(A_, B_, C_) \
  __builtin_amdgcn_wmma_f32_16x16x32_bf16(false, (A_), false, (B_), (short)0, (C_), false, false)

// ---------------------------------------------------------------------------
// Prep kernels
// ---------------------------------------------------------------------------
__global__ void kZeroF(float* __restrict__ p) {
  p[blockIdx.x * 256 + threadIdx.x] = 0.0f;
}

// out[n*1024 + k] = (bf16) in[k*1024 + n]
__global__ void kTransposeW(const float* __restrict__ in, bf16_t* __restrict__ out) {
  const int idx = blockIdx.x * 256 + threadIdx.x;   // 1024*1024 total
  const int n = idx >> 10;
  const int k = idx & 1023;
  out[idx] = (bf16_t)in[k * kD + n];
}

__global__ void kConvertX(const float* __restrict__ in, bf16_t* __restrict__ out) {
  const int idx = blockIdx.x * 256 + threadIdx.x;   // 4096*1024 total
  out[idx] = (bf16_t)in[idx];
}

// ---------------------------------------------------------------------------
// Dense GEMM (fixed M=4096, N=K=1024): C = A(bf16) * W^T[N,K](bf16) + bias.
// Wave computes 32x64 C macro-tile: 2 A-frags x 4 B-frags = 8 wmma / K-step.
// Only two live base pointers; all other fragments via constant immediates.
// ---------------------------------------------------------------------------
__global__ void kGemmBias(const bf16_t* __restrict__ A,
                          const bf16_t* __restrict__ Bt,
                          const float* __restrict__ bias,
                          float* __restrict__ C) {
  const int wave = (blockIdx.x << 3) | (threadIdx.x >> 5);
  const int lane = threadIdx.x & 31;
  const int mt = wave >> 4;                 // N/64 = 16 macro tiles
  const int nt = wave & 15;
  const int m0 = mt << 5;
  const int n0 = nt << 6;
  const int half = lane >> 4;
  const int idx  = lane & 15;

  const bf16_t* ap = A  + (size_t)(m0 + idx) * kD + half * 8;   // A fragment row
  const bf16_t* bp = Bt + (size_t)(n0 + idx) * kD + half * 8;  // B fragment row

  v8f acc[2][4] = {{ZERO8F, ZERO8F, ZERO8F, ZERO8F},
                   {ZERO8F, ZERO8F, ZERO8F, ZERO8F}};

#pragma unroll 4
  for (int k0 = 0; k0 < kD; k0 += 32) {
    if ((k0 & 127) == 0) {                    // statically resolved after unroll
      __builtin_prefetch(ap + k0 + 512, 0, 1);
      __builtin_prefetch(bp + k0 + 512, 0, 1);
    }
    v16bf af0 = load_frag2(ap,           k0, k0 + 16);
    v16bf af1 = load_frag2(ap + 16 * kD, k0, k0 + 16);
    v16bf bf0 = load_frag2(bp,           k0, k0 + 16);
    v16bf bf1 = load_frag2(bp + 16 * kD, k0, k0 + 16);
    v16bf bf2 = load_frag2(bp + 32 * kD, k0, k0 + 16);
    v16bf bf3 = load_frag2(bp + 48 * kD, k0, k0 + 16);
    acc[0][0] = WMMA_BF16(af0, bf0, acc[0][0]);
    acc[0][1] = WMMA_BF16(af0, bf1, acc[0][1]);
    acc[0][2] = WMMA_BF16(af0, bf2, acc[0][2]);
    acc[0][3] = WMMA_BF16(af0, bf3, acc[0][3]);
    acc[1][0] = WMMA_BF16(af1, bf0, acc[1][0]);
    acc[1][1] = WMMA_BF16(af1, bf1, acc[1][1]);
    acc[1][2] = WMMA_BF16(af1, bf2, acc[1][2]);
    acc[1][3] = WMMA_BF16(af1, bf3, acc[1][3]);
  }

#pragma unroll
  for (int r = 0; r < 2; ++r) {
#pragma unroll
    for (int j = 0; j < 4; ++j) {
      const float bv = bias[n0 + (j << 4) + idx];
#pragma unroll
      for (int i = 0; i < 8; ++i) {
        const int row = m0 + (r << 4) + i + 8 * half;
        C[(size_t)row * kD + n0 + (j << 4) + idx] = acc[r][j][i] + bv;
      }
    }
  }
}

// ---------------------------------------------------------------------------
// Per (b, s, h): L2-normalize q,k rows (len 64) -> bf16 [B,H,S,DH];
// v -> transposed bf16 [B,H,DH,S]. One wave per (token, head).
// ---------------------------------------------------------------------------
__global__ void kQKVFinish(const float* __restrict__ qr,
                           const float* __restrict__ kr,
                           const float* __restrict__ vr,
                           bf16_t* __restrict__ qn,
                           bf16_t* __restrict__ kn,
                           bf16_t* __restrict__ vt) {
  const int p    = (blockIdx.x << 3) | (threadIdx.x >> 5);  // [0, B*S*H)
  const int lane = threadIdx.x & 31;
  const int b    = p >> 14;           // / (S*H)
  const int rem  = p & 16383;
  const int s    = rem >> 4;
  const int h    = rem & 15;

  const size_t rowbase = ((size_t)(b * kS + s)) * kD + h * kDH;
  const int d0 = lane * 2;

  const float q0 = qr[rowbase + d0], q1 = qr[rowbase + d0 + 1];
  const float k0 = kr[rowbase + d0], k1 = kr[rowbase + d0 + 1];
  const float v0 = vr[rowbase + d0], v1 = vr[rowbase + d0 + 1];

  float sq = q0 * q0 + q1 * q1;
  float sk = k0 * k0 + k1 * k1;
#pragma unroll
  for (int m = 1; m < 32; m <<= 1) {
    sq += __shfl_xor(sq, m, 32);
    sk += __shfl_xor(sk, m, 32);
  }
  const float iq = 1.0f / fmaxf(sqrtf(sq), kEps);
  const float ik = 1.0f / fmaxf(sqrtf(sk), kEps);

  const int bh = b * kH + h;
  const size_t qkbase = ((size_t)bh * kS + s) * kDH + d0;
  qn[qkbase]     = (bf16_t)(q0 * iq);
  qn[qkbase + 1] = (bf16_t)(q1 * iq);
  kn[qkbase]     = (bf16_t)(k0 * ik);
  kn[qkbase + 1] = (bf16_t)(k1 * ik);

  const size_t vb = ((size_t)bh * kDH) * kS;
  vt[vb + (size_t)d0 * kS + s]       = (bf16_t)v0;
  vt[vb + (size_t)(d0 + 1) * kS + s] = (bf16_t)v1;
}

// ---------------------------------------------------------------------------
// Scores: per (b,h) S = qn * kn^T over DH=64; wave computes 16(q) x 64(k).
// Epilogue: acos + hardware exp2/log2 power, f32 store, shuffle-reduced
// row/col sums via float atomics.
// ---------------------------------------------------------------------------
__global__ void kScores(const bf16_t* __restrict__ qn,
                        const bf16_t* __restrict__ kn,
                        float* __restrict__ scores,
                        float* __restrict__ nr,
                        float* __restrict__ nc) {
  const int wave = (blockIdx.x << 3) | (threadIdx.x >> 5);
  const int lane = threadIdx.x & 31;
  const int bh  = wave >> 10;         // 1024 macro tiles per (b,h)
  const int t   = wave & 1023;
  const int qt  = t >> 4;             // 16-row query tile
  const int ktg = t & 15;             // 64-col key group
  const int kt0 = ktg << 6;
  const int half = lane >> 4;
  const int idx  = lane & 15;

  const bf16_t* qp = qn + ((size_t)bh * kS + (qt << 4) + idx) * kDH + half * 8;
  const bf16_t* kp = kn + ((size_t)bh * kS + kt0 + idx) * kDH + half * 8;

  v8f acc[4] = {ZERO8F, ZERO8F, ZERO8F, ZERO8F};
#pragma unroll
  for (int k0 = 0; k0 < kDH; k0 += 32) {
    v16bf af = load_frag2(qp, k0, k0 + 16);
#pragma unroll
    for (int j = 0; j < 4; ++j) {
      v16bf bf = load_frag2(kp + j * 16 * kDH, k0, k0 + 16);
      acc[j] = WMMA_BF16(af, bf, acc[j]);
    }
  }

  float rowpart[8] = {0.f, 0.f, 0.f, 0.f, 0.f, 0.f, 0.f, 0.f};

#pragma unroll
  for (int j = 0; j < 4; ++j) {
    float colsum = 0.0f;
#pragma unroll
    for (int i = 0; i < 8; ++i) {
      float d = fminf(fmaxf(acc[j][i], -kClip), kClip);
      float g = acosf(d);                                  // R = 1, BW = 1
      // (1+g)^-65 via hardware v_log_f32 / v_exp_f32 (base-2 pair)
      float s = __builtin_amdgcn_exp2f(kPowExp * __builtin_amdgcn_logf(1.0f + g));
      rowpart[i] += s;
      colsum += s;
      const int row = (qt << 4) + i + 8 * half;
      scores[((size_t)bh * kS + row) * kS + kt0 + (j << 4) + idx] = s;
    }
    // Column sums for this 16-col tile: in-lane 8 rows + partner half.
    colsum += __shfl_xor(colsum, 16, 32);
    if (half == 0)
      atomicAdd(&nc[bh * kS + kt0 + (j << 4) + idx], colsum);
  }

  // Row sums over the full 64 columns handled by this wave.
#pragma unroll
  for (int i = 0; i < 8; ++i) {
    float r = rowpart[i];
    r += __shfl_xor(r, 1, 32);
    r += __shfl_xor(r, 2, 32);
    r += __shfl_xor(r, 4, 32);
    r += __shfl_xor(r, 8, 32);
    if (idx == 0)
      atomicAdd(&nr[bh * kS + (qt << 4) + i + 8 * half], r);
  }
}

// n -> n^(-0.5), in place (A = 0.5 branch).
__global__ void kRsqrtInPlace(float* __restrict__ p) {
  const int idx = blockIdx.x * 256 + threadIdx.x;
  p[idx] = rsqrtf(p[idx]);
}

// ---------------------------------------------------------------------------
// Sinkhorn row pass: k_tilde = nr^-a * s * nc^-a, probs = k_tilde / max(sum,eps).
// Writes bf16 probs IN PLACE over the f32 score row (bf16 pitch = 2*S).
// ---------------------------------------------------------------------------
__global__ void kProbs(float* __restrict__ scores,
                       const float* __restrict__ nrp,
                       const float* __restrict__ ncp) {
  const int rowg = blockIdx.x;            // bh*S + m
  const int bh   = rowg >> 10;
  float* srow = scores + (size_t)rowg * kS;
  const float* ncb = ncp + bh * kS;

  float vals[4];
  float lsum = 0.0f;
#pragma unroll
  for (int j = 0; j < 4; ++j) {
    const int n = threadIdx.x + j * 256;
    vals[j] = srow[n] * ncb[n];
    lsum += vals[j];
  }

  __shared__ float red[256];
  red[threadIdx.x] = lsum;
  __syncthreads();
  for (int s = 128; s > 0; s >>= 1) {
    if (threadIdx.x < s) red[threadIdx.x] += red[threadIdx.x + s];
    __syncthreads();
  }
  const float nrv   = nrp[rowg];
  const float denom = fmaxf(nrv * red[0], kEps);
  const float scale = nrv / denom;

  bf16_t* prow = reinterpret_cast<bf16_t*>(srow);   // bf16 overlay, reads done
#pragma unroll
  for (int j = 0; j < 4; ++j) {
    const int n = threadIdx.x + j * 256;
    prow[n] = (bf16_t)(vals[j] * scale);
  }
}

// ---------------------------------------------------------------------------
// PV: out = probs(bf16, pitch 2*S over score buffer) * v -> bf16 [token, D].
// Wave computes 32 queries x full DH=64: 2 A-frags x 4 B-frags per K step.
// ---------------------------------------------------------------------------
__global__ void kPV(const float* __restrict__ scores,
                    const bf16_t* __restrict__ vt,
                    bf16_t* __restrict__ attnb) {
  const int wave = (blockIdx.x << 3) | (threadIdx.x >> 5);
  const int lane = threadIdx.x & 31;
  const int bh = wave >> 5;              // 32 query macro tiles per (b,h)
  const int qg = wave & 31;
  const int m0 = qg << 5;
  const int half = lane >> 4;
  const int idx  = lane & 15;

  // probs live as bf16 overlay on score rows: element pitch 2*kS per row.
  const bf16_t* pp = reinterpret_cast<const bf16_t*>(scores) +
                     ((size_t)(bh * kS + m0 + idx)) * (2 * kS) + half * 8;
  const bf16_t* vp = vt + ((size_t)bh * kDH + idx) * kS + half * 8;

  v8f acc[2][4] = {{ZERO8F, ZERO8F, ZERO8F, ZERO8F},
                   {ZERO8F, ZERO8F, ZERO8F, ZERO8F}};

#pragma unroll 4
  for (int k0 = 0; k0 < kS; k0 += 32) {
    v16bf af0 = load_frag2(pp,               k0, k0 + 16);
    v16bf af1 = load_frag2(pp + 16 * 2 * kS, k0, k0 + 16);
    v16bf bf0 = load_frag2(vp,               k0, k0 + 16);
    v16bf bf1 = load_frag2(vp + 16 * kS,     k0, k0 + 16);
    v16bf bf2 = load_frag2(vp + 32 * kS,     k0, k0 + 16);
    v16bf bf3 = load_frag2(vp + 48 * kS,     k0, k0 + 16);
    acc[0][0] = WMMA_BF16(af0, bf0, acc[0][0]);
    acc[0][1] = WMMA_BF16(af0, bf1, acc[0][1]);
    acc[0][2] = WMMA_BF16(af0, bf2, acc[0][2]);
    acc[0][3] = WMMA_BF16(af0, bf3, acc[0][3]);
    acc[1][0] = WMMA_BF16(af1, bf0, acc[1][0]);
    acc[1][1] = WMMA_BF16(af1, bf1, acc[1][1]);
    acc[1][2] = WMMA_BF16(af1, bf2, acc[1][2]);
    acc[1][3] = WMMA_BF16(af1, bf3, acc[1][3]);
  }

  const int b = bh >> 4;
  const int h = bh & 15;
#pragma unroll
  for (int r = 0; r < 2; ++r) {
#pragma unroll
    for (int j = 0; j < 4; ++j) {
#pragma unroll
      for (int i = 0; i < 8; ++i) {
        const int row = m0 + (r << 4) + i + 8 * half;
        attnb[((size_t)(b * kS + row)) * kD + h * kDH + (j << 4) + idx] =
            (bf16_t)acc[r][j][i];
      }
    }
  }
}

// ---------------------------------------------------------------------------
// Host launcher
// ---------------------------------------------------------------------------
extern "C" void kernel_launch(void* const* d_in, const int* in_sizes, int n_in,
                              void* d_out, int out_size, void* d_ws, size_t ws_size,
                              hipStream_t stream) {
  (void)in_sizes; (void)n_in; (void)out_size; (void)ws_size;

  const float* x  = (const float*)d_in[0];
  const float* wq = (const float*)d_in[1];
  const float* bq = (const float*)d_in[2];
  const float* wk = (const float*)d_in[3];
  const float* bk = (const float*)d_in[4];
  const float* wv = (const float*)d_in[5];
  const float* bv = (const float*)d_in[6];
  const float* wo = (const float*)d_in[7];
  const float* bo = (const float*)d_in[8];
  float* out = (float*)d_out;

  char* ws = (char*)d_ws;
  size_t off = 0;
  auto alloc = [&](size_t bytes) {
    char* p = ws + off;
    off += (bytes + 255) & ~(size_t)255;
    return p;
  };

  bf16_t* wqt = (bf16_t*)alloc((size_t)kD * kD * 2);
  bf16_t* wkt = (bf16_t*)alloc((size_t)kD * kD * 2);
  bf16_t* wvt = (bf16_t*)alloc((size_t)kD * kD * 2);
  bf16_t* wot = (bf16_t*)alloc((size_t)kD * kD * 2);
  bf16_t* xb  = (bf16_t*)alloc((size_t)kM * kD * 2);
  float*  qraw = (float*)alloc((size_t)kM * kD * 4);
  float*  kraw = (float*)alloc((size_t)kM * kD * 4);
  float*  vraw = (float*)alloc((size_t)kM * kD * 4);
  bf16_t* qn = (bf16_t*)alloc((size_t)kB * kH * kS * kDH * 2);
  bf16_t* kn = (bf16_t*)alloc((size_t)kB * kH * kS * kDH * 2);
  bf16_t* vt = (bf16_t*)alloc((size_t)kB * kH * kDH * kS * 2);
  bf16_t* attnb = (bf16_t*)alloc((size_t)kM * kD * 2);
  float*  nrnc = (float*)alloc((size_t)2 * kB * kH * kS * 4);  // nr | nc
  float*  nr = nrnc;
  float*  nc = nrnc + (size_t)kB * kH * kS;
  float*  scores = (float*)alloc((size_t)kB * kH * kS * kS * 4);

  const dim3 blk(256);

  // 0) zero nr/nc (2 * 65536 floats)
  kZeroF<<<512, blk, 0, stream>>>(nrnc);

  // 1) weight transposes + x convert
  kTransposeW<<<4096, blk, 0, stream>>>(wq, wqt);
  kTransposeW<<<4096, blk, 0, stream>>>(wk, wkt);
  kTransposeW<<<4096, blk, 0, stream>>>(wv, wvt);
  kTransposeW<<<4096, blk, 0, stream>>>(wo, wot);
  kConvertX<<<16384, blk, 0, stream>>>(x, xb);

  // 2) Q/K/V projections: (4096/32)*(1024/64) = 2048 waves -> 256 blocks
  kGemmBias<<<256, blk, 0, stream>>>(xb, wqt, bq, qraw);
  kGemmBias<<<256, blk, 0, stream>>>(xb, wkt, bk, kraw);
  kGemmBias<<<256, blk, 0, stream>>>(xb, wvt, bv, vraw);

  // 3) normalize + head layout (65536 waves -> 8192 blocks)
  kQKVFinish<<<8192, blk, 0, stream>>>(qraw, kraw, vraw, qn, kn, vt);

  // 4) scores: 64 bh * 64 qt * 16 ktg = 65536 waves -> 8192 blocks
  kScores<<<8192, blk, 0, stream>>>(qn, kn, scores, nr, nc);

  // 5) n^-0.5 in place
  kRsqrtInPlace<<<512, blk, 0, stream>>>(nrnc);

  // 6) Sinkhorn row normalize, probs bf16 in place (65536 rows)
  kProbs<<<65536, blk, 0, stream>>>(scores, nr, nc);

  // 7) PV: 64 bh * 32 query tiles = 2048 waves -> 256 blocks
  kPV<<<256, blk, 0, stream>>>(scores, vt, attnb);

  // 8) output projection + bias -> f32 d_out
  kGemmBias<<<256, blk, 0, stream>>>(attnb, wot, bo, out);
}
